// G_synthesis_stylegan2_84774064488479
// MI455X (gfx1250) — matrix operand order repaired and checked by reading, not compiled, not run on hardware
//
#include <hip/hip_runtime.h>
#include <hip/hip_bf16.h>

typedef __attribute__((ext_vector_type(16))) _Float16 v16h;
typedef __attribute__((ext_vector_type(8)))  _Float16 v8h;
typedef __attribute__((ext_vector_type(8)))  float    v8f;

#define LRELU(v) ((((v) < 0.0f) ? 0.2f * (v) : (v)) * 1.4142135623730951f)

// ---------------- small VALU kernels ----------------

__global__ void k_bcast(const float* __restrict__ src, float* __restrict__ dst,
                        int n, int total) {
  int t = blockIdx.x * blockDim.x + threadIdx.x;
  if (t < total) dst[t] = src[t % n];
}

__global__ void k_zeroh(_Float16* __restrict__ p, int n) {
  int t = blockIdx.x * blockDim.x + threadIdx.x;
  if (t < n) p[t] = (_Float16)0.0f;
}

// style[b,i] = lrelu( (dlat[b,:] . dw[i,:]) / sqrt(512) + db[i] ) * sqrt(2)
__global__ void k_style(const float* __restrict__ z, int zstride,
                        const float* __restrict__ dw, const float* __restrict__ db,
                        float* __restrict__ style, int B, int Cin) {
  int t = blockIdx.x * blockDim.x + threadIdx.x;
  if (t >= B * Cin) return;
  int b = t / Cin, i = t - b * Cin;
  const float g = 0.044194173824159216f;  // 1/sqrt(512)
  const float* zp = z + (size_t)b * zstride;
  const float* wp = dw + (size_t)i * 512;
  float acc = 0.f;
  for (int j = 0; j < 512; ++j) acc = fmaf(zp[j], wp[j], acc);
  float v = acc * g + db[i];
  style[t] = LRELU(v);
}

// wsq[o,i] = sum_t w[o,i,t]^2
__global__ void k_wsq(const float* __restrict__ w, float* __restrict__ wsq,
                      int Cout, int Cin) {
  int t = blockIdx.x * blockDim.x + threadIdx.x;
  if (t >= Cout * Cin) return;
  const float* wp = w + (size_t)t * 9;
  float s = 0.f;
  for (int k = 0; k < 9; ++k) s = fmaf(wp[k], wp[k], s);
  wsq[t] = s;
}

// demod[b,o] = rsqrt( scale^2 * sum_i style[b,i]^2 * wsq[o,i] + 1e-8 )
__global__ void k_demod(const float* __restrict__ style, const float* __restrict__ wsq,
                        float* __restrict__ demod, int B, int Cout, int Cin) {
  int t = blockIdx.x * blockDim.x + threadIdx.x;
  if (t >= B * Cout) return;
  int b = t / Cout, o = t - b * Cout;
  const float sc2 = 1.0f / (float)(Cin * 9);
  const float* sp = style + (size_t)b * Cin;
  const float* wp = wsq + (size_t)o * Cin;
  float s = 0.f;
  for (int i = 0; i < Cin; ++i) { float si = sp[i]; s = fmaf(si * si, wp[i], s); }
  demod[t] = rsqrtf(sc2 * s + 1e-8f);
}

// Pack activations: NCHW f32 -> NHWC f16, modulated by style, optional 1-px
// zero border (pad=1 for the 3x3 same-conv; pad=0 for the transposed conv).
__global__ void k_packx(const float* __restrict__ x, const float* __restrict__ style,
                        _Float16* __restrict__ xh, int B, int Cin, int H, int pad) {
  int Hp = H + 2 * pad;
  int t = blockIdx.x * blockDim.x + threadIdx.x;
  int total = B * Hp * Hp * Cin;
  if (t >= total) return;
  int c = t % Cin; int r = t / Cin;
  int xx = r % Hp - pad; r /= Hp;
  int yy = r % Hp - pad; int b = r / Hp;
  float v = 0.f;
  if (yy >= 0 && yy < H && xx >= 0 && xx < H)
    v = x[(((size_t)b * Cin + c) * H + yy) * H + xx] * style[b * Cin + c];
  xh[t] = (_Float16)v;
}

// Pack weights: [Cout,Cin,3,3] f32 -> [tap][Cout][Cin] f16 * (1/sqrt(Cin*9)),
// with optional spatial flip (tap -> 8-tap) for the transposed conv.
__global__ void k_packw(const float* __restrict__ w, _Float16* __restrict__ wh,
                        int Cout, int Cin, int flip) {
  int t = blockIdx.x * blockDim.x + threadIdx.x;
  int total = 9 * Cout * Cin;
  if (t >= total) return;
  int c = t % Cin; int r = t / Cin;
  int o = r % Cout; int tap = r / Cout;
  int tsrc = flip ? (8 - tap) : tap;
  float scale = rsqrtf((float)(Cin * 9));
  wh[t] = (_Float16)(w[((size_t)o * Cin + c) * 9 + tsrc] * scale);
}

// ---------------- WMMA implicit-GEMM convolutions ----------------
// One wave computes 4 M-tiles (64 pixels) x 2 N-tiles (32 outch): each K-chunk
// issues 4 B-loads + 8 A-loads (b128) for 8 v_wmma_f32_16x16x32_f16, reusing
// each A fragment across both N tiles and each B fragment across all 4 M
// tiles. Invalid rows read a zeroed guard block via branch-free address
// select (EXEC stays all-ones at every WMMA).

__global__ __launch_bounds__(32) void k_conv3x3_wmma(
    const _Float16* __restrict__ xh,   // [B, H+2, H+2, Cin] padded, modulated
    const _Float16* __restrict__ wh,   // [9, Cout, Cin]
    const _Float16* __restrict__ zbuf, // >= Cin+32 zero f16
    const float* __restrict__ demod, const float* __restrict__ noise,
    const float* __restrict__ nstr, const float* __restrict__ bias,
    float* __restrict__ out, int Cin, int Cout, int H) {
  const int lane = threadIdx.x;
  const int half = (lane >> 4) & 1;
  const int l16 = lane & 15;
  const int mt0 = blockIdx.x * 4, nt = blockIdx.y, b = blockIdx.z;
  const int npix = H * H;
  const int Hp = H + 2;
  const int n0 = nt * 32 + l16;        // first N tile; second is n0+16
  const _Float16* xb = xh + (size_t)b * Hp * Hp * Cin;

  int py[4], px[4], mv[4];
#pragma unroll
  for (int s = 0; s < 4; ++s) {
    int m = (mt0 + s) * 16 + l16;
    mv[s] = (m < npix);
    int mm = mv[s] ? m : 0;
    py[s] = mm / H; px[s] = mm - py[s] * H;
  }

  v8f acc0[4] = {{}, {}, {}, {}};
  v8f acc1[4] = {{}, {}, {}, {}};
  for (int tap = 0; tap < 9; ++tap) {
    const int dy = tap / 3 - 1, dx = tap - (tap / 3) * 3 - 1;
    const _Float16* wt = wh + ((size_t)tap * Cout + n0) * Cin + half * 16;
    const _Float16* rowA[4];
#pragma unroll
    for (int s = 0; s < 4; ++s)
      rowA[s] = mv[s]
          ? xb + ((size_t)(py[s] + 1 + dy) * Hp + (px[s] + 1 + dx)) * Cin + half * 16
          : zbuf;
    for (int c0 = 0; c0 < Cin; c0 += 32) {
      v16h B0 = *(const v16h*)(wt + c0);               // n0 tile
      v16h B1 = *(const v16h*)(wt + (size_t)16 * Cin + c0);  // n0+16 tile
#pragma unroll
      for (int s = 0; s < 4; ++s) {
        union { v16h v; v8h h[2]; } A;
        A.h[0] = *(const v8h*)(rowA[s] + c0);        // k = half*16 + 0..7
        A.h[1] = *(const v8h*)(rowA[s] + c0 + 8);    // k = half*16 + 8..15
        acc0[s] = __builtin_amdgcn_wmma_f32_16x16x32_f16(
            false, A.v, false, B0, (short)0, acc0[s], false, false);
        acc1[s] = __builtin_amdgcn_wmma_f32_16x16x32_f16(
            false, A.v, false, B1, (short)0, acc1[s], false, false);
      }
    }
  }

  const float dm0 = demod[b * Cout + n0];
  const float dm1 = demod[b * Cout + n0 + 16];
  const float bs0 = bias[n0], bs1 = bias[n0 + 16];
  const float ns = nstr[0];
#pragma unroll
  for (int s = 0; s < 4; ++s) {
    if ((mt0 + s) * 16 >= npix) continue;  // npix is a multiple of 16
#pragma unroll
    for (int r = 0; r < 8; ++r) {
      int mm = (mt0 + s) * 16 + r + half * 8;
      float nz = noise[mm] * ns;
      float v0 = fmaf(acc0[s][r], dm0, nz) + bs0;
      float v1 = fmaf(acc1[s][r], dm1, nz) + bs1;
      out[((size_t)b * Cout + n0) * npix + mm] = LRELU(v0);
      out[((size_t)b * Cout + n0 + 16) * npix + mm] = LRELU(v1);
    }
  }
}

// Transposed conv (stride 2, k=3): output (2H+1)^2, demod applied, raw store.
// Parity/bounds zero-fill via guard-block address select.
__global__ __launch_bounds__(32) void k_convT3x3_wmma(
    const _Float16* __restrict__ xh,   // [B, Hin, Hin, Cin] modulated
    const _Float16* __restrict__ wh,   // [9, Cout, Cin], taps pre-flipped
    const _Float16* __restrict__ zbuf,
    const float* __restrict__ demod, float* __restrict__ out,
    int Cin, int Cout, int Hin) {
  const int Ho = 2 * Hin + 1;
  const int npix = Ho * Ho;
  const int lane = threadIdx.x;
  const int half = (lane >> 4) & 1;
  const int l16 = lane & 15;
  const int mt0 = blockIdx.x * 4, nt = blockIdx.y, b = blockIdx.z;
  const int n0 = nt * 32 + l16;
  const _Float16* xb = xh + (size_t)b * Hin * Hin * Cin;

  int Y[4], X[4], mv[4];
#pragma unroll
  for (int s = 0; s < 4; ++s) {
    int m = (mt0 + s) * 16 + l16;
    mv[s] = (m < npix);
    int mm = mv[s] ? m : 0;
    Y[s] = mm / Ho; X[s] = mm - Y[s] * Ho;
  }

  v8f acc0[4] = {{}, {}, {}, {}};
  v8f acc1[4] = {{}, {}, {}, {}};
  for (int tap = 0; tap < 9; ++tap) {
    const int ky = tap / 3, kx = tap - (tap / 3) * 3;
    const _Float16* wt = wh + ((size_t)tap * Cout + n0) * Cin + half * 16;
    const _Float16* rowA[4];
#pragma unroll
    for (int s = 0; s < 4; ++s) {
      int ty = Y[s] + ky - 2, tx = X[s] + kx - 2;
      bool ok = mv[s] && ty >= 0 && tx >= 0 && !(ty & 1) && !(tx & 1) &&
                (ty >> 1) < Hin && (tx >> 1) < Hin;
      rowA[s] = ok ? xb + ((size_t)(ty >> 1) * Hin + (tx >> 1)) * Cin + half * 16
                   : zbuf;
    }
    for (int c0 = 0; c0 < Cin; c0 += 32) {
      v16h B0 = *(const v16h*)(wt + c0);
      v16h B1 = *(const v16h*)(wt + (size_t)16 * Cin + c0);
#pragma unroll
      for (int s = 0; s < 4; ++s) {
        union { v16h v; v8h h[2]; } A;
        A.h[0] = *(const v8h*)(rowA[s] + c0);
        A.h[1] = *(const v8h*)(rowA[s] + c0 + 8);
        acc0[s] = __builtin_amdgcn_wmma_f32_16x16x32_f16(
            false, A.v, false, B0, (short)0, acc0[s], false, false);
        acc1[s] = __builtin_amdgcn_wmma_f32_16x16x32_f16(
            false, A.v, false, B1, (short)0, acc1[s], false, false);
      }
    }
  }

  const float dm0 = demod[b * Cout + n0];
  const float dm1 = demod[b * Cout + n0 + 16];
#pragma unroll
  for (int s = 0; s < 4; ++s) {
#pragma unroll
    for (int r = 0; r < 8; ++r) {
      int mm = (mt0 + s) * 16 + r + half * 8;
      if (mm < npix) {
        out[((size_t)b * Cout + n0) * npix + mm] = acc0[s][r] * dm0;
        out[((size_t)b * Cout + n0 + 16) * npix + mm] = acc1[s][r] * dm1;
      }
    }
  }
}

// 4x4 FIR after transposed conv: (2h+1)^2 -> (2h)^2, then noise+bias+lrelu.
__global__ void k_fir_noise_act(const float* __restrict__ yT,
                                const float* __restrict__ noise,
                                const float* __restrict__ nstr,
                                const float* __restrict__ bias,
                                float* __restrict__ out, int B, int C, int Hout) {
  int t = blockIdx.x * blockDim.x + threadIdx.x;
  int total = B * C * Hout * Hout;
  if (t >= total) return;
  int X = t % Hout; int r = t / Hout;
  int Y = r % Hout; r /= Hout;
  int c = r % C; int b = r / C;
  const int Hin = Hout + 1;
  const float cf[4] = {1.f, 3.f, 3.f, 1.f};
  const float* src = yT + (size_t)(b * C + c) * Hin * Hin;
  float s = 0.f;
  for (int a = 0; a < 4; ++a) {
    int yy = Y + a - 1;
    if (yy < 0 || yy >= Hin) continue;
    float rowacc = 0.f;
    for (int bb = 0; bb < 4; ++bb) {
      int xx = X + bb - 1;
      if (xx < 0 || xx >= Hin) continue;
      rowacc = fmaf(cf[bb], src[yy * Hin + xx], rowacc);
    }
    s = fmaf(cf[a], rowacc, s);
  }
  s *= 0.0625f;
  s = fmaf(noise[Y * Hout + X], nstr[0], s) + bias[c];
  out[t] = LRELU(s);
}

// 1x1 modulated conv to RGB (no demod) + bias.
__global__ void k_torgb(const float* __restrict__ x, const float* __restrict__ w,
                        const float* __restrict__ style, const float* __restrict__ bias,
                        float* __restrict__ rgb, int B, int Cin, int H) {
  int t = blockIdx.x * blockDim.x + threadIdx.x;
  int npix = H * H;
  if (t >= B * npix) return;
  int b = t / npix, p = t - b * npix;
  const float scale = rsqrtf((float)Cin);
  const float* xp = x + (size_t)b * Cin * npix + p;
  const float* sp = style + (size_t)b * Cin;
  float a0 = 0.f, a1 = 0.f, a2 = 0.f;
  for (int i = 0; i < Cin; ++i) {
    float xv = xp[(size_t)i * npix] * sp[i];
    a0 = fmaf(w[i], xv, a0);
    a1 = fmaf(w[Cin + i], xv, a1);
    a2 = fmaf(w[2 * Cin + i], xv, a2);
  }
  float* op = rgb + (size_t)b * 3 * npix + p;
  op[0] = a0 * scale + bias[0];
  op[(size_t)npix] = a1 * scale + bias[1];
  op[(size_t)2 * npix] = a2 * scale + bias[2];
}

// skip_out = rgb + upfirdn(skip_prev, up=2, pad(2,1,2,1), coeff c_a*c_b/16)
__global__ void k_rgb_skip(const float* __restrict__ rgb,
                           const float* __restrict__ prev,
                           float* __restrict__ out, int B, int H) {
  int t = blockIdx.x * blockDim.x + threadIdx.x;
  int npix = H * H;
  if (t >= B * 3 * npix) return;
  float v = rgb[t];
  if (prev) {
    int X = t % H; int r = t / H;
    int Y = r % H; int ch = r / H;
    const int Hp = H >> 1;
    const float cf[4] = {1.f, 3.f, 3.f, 1.f};
    const float* sp = prev + (size_t)ch * Hp * Hp;
    float s = 0.f;
    for (int a = 0; a < 4; ++a) {
      int dyc = Y + a - 2;
      if (dyc < 0 || (dyc & 1) || (dyc >> 1) >= Hp) continue;
      float rowacc = 0.f;
      for (int bb = 0; bb < 4; ++bb) {
        int dxc = X + bb - 2;
        if (dxc < 0 || (dxc & 1) || (dxc >> 1) >= Hp) continue;
        rowacc = fmaf(cf[bb], sp[(dyc >> 1) * Hp + (dxc >> 1)], rowacc);
      }
      s = fmaf(cf[a], rowacc, s);
    }
    v += s * 0.0625f;
  }
  out[t] = v;
}

// ---------------- host orchestration ----------------

extern "C" void kernel_launch(void* const* d_in, const int* in_sizes, int n_in,
                              void* d_out, int out_size, void* d_ws, size_t ws_size,
                              hipStream_t stream) {
  (void)in_sizes; (void)n_in; (void)out_size; (void)ws_size;
  const int B = 4;
  static const int NFv[7]  = {512, 512, 512, 512, 512, 256, 128};
  static const int RESv[7] = {4, 8, 16, 32, 64, 128, 256};

  int idx = 0;
  const float* dlat = (const float*)d_in[idx++];  // [4,14,512]
  const float* cin0 = (const float*)d_in[idx++];  // [1,512,4,4]
  struct ConvP { const float *w, *dw, *db, *bias, *nstr, *nconst; };
  struct RgbP  { const float *w, *dw, *db, *bias; };
  ConvP ups[6], mains[7]; RgbP trgbs[7];
  for (int i = 0; i < 6; ++i) {
    ups[i].w = (const float*)d_in[idx++]; ups[i].dw = (const float*)d_in[idx++];
    ups[i].db = (const float*)d_in[idx++]; ups[i].bias = (const float*)d_in[idx++];
    ups[i].nstr = (const float*)d_in[idx++]; ups[i].nconst = (const float*)d_in[idx++];
  }
  for (int i = 0; i < 7; ++i) {
    mains[i].w = (const float*)d_in[idx++]; mains[i].dw = (const float*)d_in[idx++];
    mains[i].db = (const float*)d_in[idx++]; mains[i].bias = (const float*)d_in[idx++];
    mains[i].nstr = (const float*)d_in[idx++]; mains[i].nconst = (const float*)d_in[idx++];
  }
  for (int i = 0; i < 7; ++i) {
    trgbs[i].w = (const float*)d_in[idx++]; trgbs[i].dw = (const float*)d_in[idx++];
    trgbs[i].db = (const float*)d_in[idx++]; trgbs[i].bias = (const float*)d_in[idx++];
  }

  float* wsf = (float*)d_ws;
  size_t off = 0;
  auto alloc = [&](size_t n) { float* p = wsf + off; off += n; return p; };
  const size_t BIG = 35651584;  // floats
  float* XA = alloc(BIG); float* XB = alloc(BIG); float* YT = alloc(BIG);
  _Float16* XH = (_Float16*)alloc(BIG / 2);     // 35.6M f16: 4*258*258*128 fits
  _Float16* WH = (_Float16*)alloc(1179648);     // 9*512*512 f16
  _Float16* ZH = (_Float16*)alloc(512);         // 1024 zero f16 guard block
  float* STYLE = alloc(4096); float* DEMOD = alloc(4096); float* WSQ = alloc(262144);
  float* SKA = alloc(786432); float* SKB = alloc(786432); float* RGBB = alloc(786432);

  const int DSTRIDE = 14 * 512;
  auto cdiv = [](int a, int b) { return (a + b - 1) / b; };

  k_zeroh<<<4, 256, 0, stream>>>(ZH, 1024);
  { int n = 512 * 16; int tot = n * B;  // x = tile(const_input, B)
    k_bcast<<<cdiv(tot, 256), 256, 0, stream>>>(cin0, XA, n, tot); }

  float* X = XA; float* XO = XB;
  const float* skipPrev = nullptr;
  float* skipBuf = SKA;
  int in_ch = 512;

  for (int i = 0; i < 7; ++i) {
    const int H = RESv[i], Cout = NFv[i], rl = i + 2;
    if (H > 4) {  // up layer: convT -> FIR+noise+act
      const ConvP& L = ups[i - 1];
      const int Cin = in_ch, Hin = H / 2, Ho = H + 1;
      k_style<<<cdiv(B * Cin, 256), 256, 0, stream>>>(dlat + (2 * rl - 5) * 512, DSTRIDE, L.dw, L.db, STYLE, B, Cin);
      k_wsq<<<cdiv(Cout * Cin, 256), 256, 0, stream>>>(L.w, WSQ, Cout, Cin);
      k_demod<<<cdiv(B * Cout, 256), 256, 0, stream>>>(STYLE, WSQ, DEMOD, B, Cout, Cin);
      k_packx<<<cdiv(B * Hin * Hin * Cin, 256), 256, 0, stream>>>(X, STYLE, XH, B, Cin, Hin, 0);
      k_packw<<<cdiv(9 * Cout * Cin, 256), 256, 0, stream>>>(L.w, WH, Cout, Cin, 1);
      dim3 g(cdiv(Ho * Ho, 64), Cout / 32, B);
      k_convT3x3_wmma<<<g, 32, 0, stream>>>(XH, WH, ZH, DEMOD, YT, Cin, Cout, Hin);
      int tot = B * Cout * H * H;
      k_fir_noise_act<<<cdiv(tot, 256), 256, 0, stream>>>(YT, L.nconst, L.nstr, L.bias, XO, B, Cout, H);
      float* t = X; X = XO; XO = t;
      in_ch = Cout;
    }
    {  // main layer
      const ConvP& M = mains[i];
      const int Cin = in_ch;
      k_style<<<cdiv(B * Cin, 256), 256, 0, stream>>>(dlat + (2 * rl - 4) * 512, DSTRIDE, M.dw, M.db, STYLE, B, Cin);
      k_wsq<<<cdiv(Cout * Cin, 256), 256, 0, stream>>>(M.w, WSQ, Cout, Cin);
      k_demod<<<cdiv(B * Cout, 256), 256, 0, stream>>>(STYLE, WSQ, DEMOD, B, Cout, Cin);
      k_packx<<<cdiv(B * (H + 2) * (H + 2) * Cin, 256), 256, 0, stream>>>(X, STYLE, XH, B, Cin, H, 1);
      k_packw<<<cdiv(9 * Cout * Cin, 256), 256, 0, stream>>>(M.w, WH, Cout, Cin, 0);
      dim3 g(cdiv(H * H, 64), Cout / 32, B);
      k_conv3x3_wmma<<<g, 32, 0, stream>>>(XH, WH, ZH, DEMOD, M.nconst, M.nstr, M.bias, XO, Cin, Cout, H);
      float* t = X; X = XO; XO = t;
      in_ch = Cout;
    }
    {  // toRGB + skip accumulation
      const RgbP& T = trgbs[i];
      k_style<<<cdiv(B * in_ch, 256), 256, 0, stream>>>(dlat + (2 * rl - 3) * 512, DSTRIDE, T.dw, T.db, STYLE, B, in_ch);
      k_torgb<<<cdiv(B * H * H, 256), 256, 0, stream>>>(X, T.w, STYLE, T.bias, RGBB, B, in_ch, H);
      float* dst = (i == 6) ? (float*)d_out : skipBuf;
      k_rgb_skip<<<cdiv(B * 3 * H * H, 256), 256, 0, stream>>>(RGBB, skipPrev, dst, B, H);
      skipPrev = dst;
      skipBuf = (skipBuf == SKA) ? SKB : SKA;
    }
  }
}
